// GAT_15685220565371
// MI455X (gfx1250) — matrix operand order repaired and checked by reading, not compiled, not run on hardware
//
#include <hip/hip_runtime.h>

// ---- problem constants (from reference) ----
#define NB 4
#define NN 20000
#define NT (NB * NN)          // 80000 batched nodes
#define DEG 16
#define NE (NB * NN * DEG)    // 1,280,000 edges
#define ETOT (NE + NT)        // + self loops = 1,360,000
#define HEADS 8
#define HIDC 6
#define F1 48                 // HEADS*HIDC
#define RAISED_ 8
#define POOLW 100
#define NPOOL (NN / POOLW)    // 200
#define LIN_IN (NPOOL * F1)   // 9600
#define LIN_MID (LIN_IN / 4)  // 2400
#define NCLS 5

typedef __attribute__((ext_vector_type(16))) _Float16 v16h;
typedef __attribute__((ext_vector_type(8)))  _Float16 v8h;
typedef __attribute__((ext_vector_type(8)))  float    v8f;

union V16U { v16h v; v8h h[2]; };

__device__ __forceinline__ float leakyf(float v, float s) { return v > 0.f ? v : s * v; }

__device__ __forceinline__ float atomicMaxF(float* addr, float val) {
  int* a = (int*)addr;
  int old = __float_as_int(*addr);
  while (__int_as_float(old) < val) {
    int assumed = old;
    old = atomicCAS(a, assumed, __float_as_int(val));
    if (old == assumed) break;
  }
  return __int_as_float(old);
}

__device__ __forceinline__ void edge_sd(const long long* __restrict__ ei, int e, int& s, int& d) {
  if (e < NE) { s = (int)ei[e]; d = (int)ei[(size_t)NE + e]; }
  else        { s = e - NE; d = s; }   // appended self loops
}

// ---------------- fill ----------------
__global__ void fill_kernel(float* __restrict__ p, float v, int n) {
  int t = blockIdx.x * blockDim.x + threadIdx.x;
  if (t < n) p[t] = v;
}

// ---------------- pre: h0 = relu(x @ pre_W + pre_b), x is [Nt,1] ----------------
__global__ void pre_kernel(const float* __restrict__ x, const float* __restrict__ W,
                           const float* __restrict__ b, float* __restrict__ h0) {
  int t = blockIdx.x * blockDim.x + threadIdx.x;
  if (t >= NT * RAISED_) return;
  int n = t >> 3, j = t & 7;
  float v = x[n] * W[j] + b[j];
  h0[t] = v > 0.f ? v : 0.f;
}

// ---------------- WMMA node-feature GEMM: Y[Nt,48] = X[Nt,KIN] @ W[KIN,48] ----------------
// 4 waves/block, each wave does a 16x48 tile via v_wmma_f32_16x16x32_f16.
// Operands are staged in LDS as f16, zero-padded to KP so fragment builds are
// branch-free contiguous ds_load_b128 pairs (EXEC stays all-ones around WMMA).
template <int KIN>
__global__ __launch_bounds__(128)
void gemm_wmma_kernel(const float* __restrict__ X, const float* __restrict__ Wm,
                      float* __restrict__ Y) {
  constexpr int KP = (KIN + 31) & ~31;                  // padded K: 32 or 64
  __shared__ __align__(16) _Float16 ldsW[F1 * KP];      // W transposed: [n][KP]
  __shared__ __align__(16) _Float16 ldsA[4][16 * KP];   // per-wave A tile: [row][KP]

  const int wave = threadIdx.x >> 5;
  const int lane = threadIdx.x & 31;
  const int hi   = lane >> 4;
  const int lo   = lane & 15;
  const int rowBase = (blockIdx.x * 4 + wave) * 16;     // NT % 64 == 0 -> in range

  // stage W transposed + padded (whole block)
  for (int i = threadIdx.x; i < F1 * KP; i += 128) {
    int n = i / KP, k = i % KP;
    ldsW[i] = (k < KIN) ? (_Float16)Wm[k * F1 + n] : (_Float16)0.f;
  }
  // stage this wave's 16-row A tile, padded (coalesced along k)
  for (int i = lane; i < 16 * KP; i += 32) {
    int r = i / KP, k = i % KP;
    ldsA[wave][i] = (k < KIN) ? (_Float16)X[(size_t)(rowBase + r) * KIN + k]
                              : (_Float16)0.f;
  }
  __syncthreads();

  v8f acc0 = {}, acc1 = {}, acc2 = {};
  constexpr int NCH = KP / 32;

#pragma unroll
  for (int kc = 0; kc < NCH; ++kc) {
    const int kbase = kc * 32;
    // A fragment (16-bit A 16x32 layout): lane<16 holds K {0..7,16..23},
    // lane>=16 holds K {8..15,24..31} -> two contiguous 8-half runs.
    V16U A;
    const _Float16* arow = &ldsA[wave][lo * KP + kbase];
    A.h[0] = *(const v8h*)(arow + hi * 8);
    A.h[1] = *(const v8h*)(arow + 16 + hi * 8);
#pragma unroll
    for (int nt = 0; nt < 3; ++nt) {
      // B fragment: lanes 0-15 K {0..7,16..23}, lanes 16-31 K {8..15,24..31} at fixed N.
      V16U Bf;
      const _Float16* bcol = &ldsW[(nt * 16 + lo) * KP + kbase + hi * 8];
      Bf.h[0] = *(const v8h*)(bcol);
      Bf.h[1] = *(const v8h*)(bcol + 16);
      v8f& acc = (nt == 0) ? acc0 : (nt == 1 ? acc1 : acc2);
      acc = __builtin_amdgcn_wmma_f32_16x16x32_f16(false, A.v, false, Bf.v,
                                                   (short)0, acc, false, false);
    }
  }
  // C/D layout: VGPR i holds row (i + 8*hi), col = lane&15
#pragma unroll
  for (int nt = 0; nt < 3; ++nt) {
    const v8f& acc = (nt == 0) ? acc0 : (nt == 1 ? acc1 : acc2);
#pragma unroll
    for (int i = 0; i < 8; ++i) {
      int row = rowBase + hi * 8 + i;
      Y[(size_t)row * F1 + nt * 16 + lo] = acc[i];
    }
  }
}

// ---------------- attention dot products: a_src/a_dst [Nt,HEADS] ----------------
__global__ void dots_kernel(const float* __restrict__ H, const float* __restrict__ as_,
                            const float* __restrict__ ad_, float* __restrict__ asrc,
                            float* __restrict__ adst) {
  int t = blockIdx.x * blockDim.x + threadIdx.x;
  if (t >= NT * HEADS) return;
  int n = t >> 3, h = t & 7;
  float s = 0.f, d = 0.f;
#pragma unroll
  for (int c = 0; c < HIDC; ++c) {
    float v = H[(size_t)n * F1 + h * HIDC + c];
    s += v * as_[h * HIDC + c];
    d += v * ad_[h * HIDC + c];
  }
  asrc[t] = s;
  adst[t] = d;
}

// ---------------- edge pass 1: segment max (per dst,head) ----------------
__global__ void edge_max_kernel(const long long* __restrict__ ei,
                                const float* __restrict__ asrc, const float* __restrict__ adst,
                                float* __restrict__ m) {
  int t = blockIdx.x * blockDim.x + threadIdx.x;
  if (t >= ETOT * HEADS) return;
  int e = t >> 3, h = t & 7;
  int s, d; edge_sd(ei, e, s, d);
  float v = leakyf(asrc[s * HEADS + h] + adst[d * HEADS + h], 0.2f);
  atomicMaxF(&m[d * HEADS + h], v);
}

// ---------------- edge pass 2: exp-sum ----------------
__global__ void edge_expsum_kernel(const long long* __restrict__ ei,
                                   const float* __restrict__ asrc, const float* __restrict__ adst,
                                   const float* __restrict__ m, float* __restrict__ ssum) {
  int t = blockIdx.x * blockDim.x + threadIdx.x;
  if (t >= ETOT * HEADS) return;
  int e = t >> 3, h = t & 7;
  int s, d; edge_sd(ei, e, s, d);
  float v = leakyf(asrc[s * HEADS + h] + adst[d * HEADS + h], 0.2f);
  float ex = __expf(v - m[d * HEADS + h]);
  atomicAdd(&ssum[d * HEADS + h], ex);
}

// ---------------- edge pass 3: normalized weighted scatter ----------------
__global__ void edge_scatter_kernel(const long long* __restrict__ ei,
                                    const float* __restrict__ asrc, const float* __restrict__ adst,
                                    const float* __restrict__ m, const float* __restrict__ ssum,
                                    const float* __restrict__ H, float* __restrict__ agg) {
  int t = blockIdx.x * blockDim.x + threadIdx.x;
  if (t >= ETOT * HEADS) return;
  int e = t >> 3, h = t & 7;
  int s, d; edge_sd(ei, e, s, d);
  float v = leakyf(asrc[s * HEADS + h] + adst[d * HEADS + h], 0.2f);
  float ex = __expf(v - m[d * HEADS + h]);
  float alpha = ex / (ssum[d * HEADS + h] + 1e-16f);
#pragma unroll
  for (int c = 0; c < HIDC; ++c) {
    atomicAdd(&agg[(size_t)d * F1 + h * HIDC + c],
              alpha * H[(size_t)s * F1 + h * HIDC + c]);
  }
}

// ---------------- bias + leaky(0.01) ----------------
__global__ void bias_leaky_kernel(const float* __restrict__ agg, const float* __restrict__ b,
                                  float* __restrict__ out) {
  int t = blockIdx.x * blockDim.x + threadIdx.x;
  if (t >= NT * F1) return;
  float v = agg[t] + b[t % F1];
  out[t] = leakyf(v, 0.01f);
}

// ---------------- max pool over POOLW nodes ----------------
__global__ void maxpool_kernel(const float* __restrict__ h, float* __restrict__ o) {
  int t = blockIdx.x * blockDim.x + threadIdx.x;
  if (t >= NB * NPOOL * F1) return;
  int b = t / (NPOOL * F1);
  int rem = t % (NPOOL * F1);
  int p = rem / F1, f = rem % F1;
  size_t base = ((size_t)b * NN + (size_t)p * POOLW) * F1 + f;
  float mx = -3.0e38f;
  for (int i = 0; i < POOLW; ++i) mx = fmaxf(mx, h[base + (size_t)i * F1]);
  o[t] = mx;
}

// ---------------- dense layer (M = NB = 4 rows) ----------------
__global__ void lin_kernel(const float* __restrict__ in, const float* __restrict__ W,
                           const float* __restrict__ bias, float* __restrict__ o,
                           int K, int Ncol, int relu) {
  int t = blockIdx.x * blockDim.x + threadIdx.x;
  if (t >= NB * Ncol) return;
  int r = t / Ncol, c = t % Ncol;
  float acc = bias[c];
  const float* ip = in + (size_t)r * K;
  for (int k = 0; k < K; ++k) acc += ip[k] * W[(size_t)k * Ncol + c];
  if (relu) acc = acc > 0.f ? acc : 0.f;
  o[t] = acc;
}

// ---------------- classifier + log_softmax ----------------
__global__ void clf_kernel(const float* __restrict__ in, const float* __restrict__ W,
                           const float* __restrict__ bias, float* __restrict__ o) {
  int r = threadIdx.x;
  if (r >= NB) return;
  float lg[NCLS];
  float mx = -3.0e38f;
#pragma unroll
  for (int c = 0; c < NCLS; ++c) {
    float a = bias[c];
    for (int k = 0; k < 64; ++k) a += in[r * 64 + k] * W[k * NCLS + c];
    lg[c] = a;
    mx = fmaxf(mx, a);
  }
  float s = 0.f;
#pragma unroll
  for (int c = 0; c < NCLS; ++c) s += __expf(lg[c] - mx);
  float lse = __logf(s);
#pragma unroll
  for (int c = 0; c < NCLS; ++c) o[r * NCLS + c] = lg[c] - mx - lse;
}

extern "C" void kernel_launch(void* const* d_in, const int* in_sizes, int n_in,
                              void* d_out, int out_size, void* d_ws, size_t ws_size,
                              hipStream_t stream) {
  (void)in_sizes; (void)n_in; (void)out_size; (void)ws_size;
  const float*     x     = (const float*)d_in[0];
  const long long* ei    = (const long long*)d_in[1];  // edge_index int64 [2,E]
  // d_in[2] edge_weight: unused by reference
  const float* pre_W = (const float*)d_in[3];
  const float* pre_b = (const float*)d_in[4];
  const float* W1    = (const float*)d_in[5];
  const float* as1   = (const float*)d_in[6];
  const float* ad1   = (const float*)d_in[7];
  const float* b1    = (const float*)d_in[8];
  const float* W2    = (const float*)d_in[9];
  const float* as2   = (const float*)d_in[10];
  const float* ad2   = (const float*)d_in[11];
  const float* b2    = (const float*)d_in[12];
  const float* l1W   = (const float*)d_in[13];
  const float* l1b   = (const float*)d_in[14];
  const float* l2W   = (const float*)d_in[15];
  const float* l2b   = (const float*)d_in[16];
  const float* cW    = (const float*)d_in[17];
  const float* cb    = (const float*)d_in[18];
  float* out = (float*)d_out;

  // workspace layout (floats)
  float* bufA = (float*)d_ws;                       // node features in   [Nt,48] (h0 uses [Nt,8])
  float* bufB = bufA + (size_t)NT * F1;             // transformed feats  [Nt,48]
  float* bufC = bufB + (size_t)NT * F1;             // aggregated output  [Nt,48]
  float* asrc = bufC + (size_t)NT * F1;             // [Nt,8]
  float* adst = asrc + (size_t)NT * HEADS;          // [Nt,8]
  float* mbuf = adst + (size_t)NT * HEADS;          // segment max [Nt,8]
  float* sbuf = mbuf + (size_t)NT * HEADS;          // segment sum [Nt,8]
  float* pool = sbuf + (size_t)NT * HEADS;          // [4,200,48]
  float* l1o  = pool + (size_t)NB * NPOOL * F1;     // [4,2400]
  float* l2o  = l1o  + (size_t)NB * LIN_MID;        // [4,64]

  const int TB = 256;
  auto blks = [](long long n, int tb) { return (unsigned)((n + tb - 1) / tb); };

  // ---- pre layer ----
  pre_kernel<<<blks((long long)NT * RAISED_, TB), TB, 0, stream>>>(x, pre_W, pre_b, bufA);

  const long long nNH = (long long)NT * HEADS;
  const long long nNF = (long long)NT * F1;
  const long long nEH = (long long)ETOT * HEADS;

  // ---- GATConv 1 ----
  gemm_wmma_kernel<RAISED_><<<NT / 64, 128, 0, stream>>>(bufA, W1, bufB);
  dots_kernel<<<blks(nNH, TB), TB, 0, stream>>>(bufB, as1, ad1, asrc, adst);
  fill_kernel<<<blks(nNH, TB), TB, 0, stream>>>(mbuf, -3.0e38f, (int)nNH);
  fill_kernel<<<blks(nNH, TB), TB, 0, stream>>>(sbuf, 0.f, (int)nNH);
  fill_kernel<<<blks(nNF, TB), TB, 0, stream>>>(bufC, 0.f, (int)nNF);
  edge_max_kernel<<<blks(nEH, TB), TB, 0, stream>>>(ei, asrc, adst, mbuf);
  edge_expsum_kernel<<<blks(nEH, TB), TB, 0, stream>>>(ei, asrc, adst, mbuf, sbuf);
  edge_scatter_kernel<<<blks(nEH, TB), TB, 0, stream>>>(ei, asrc, adst, mbuf, sbuf, bufB, bufC);
  bias_leaky_kernel<<<blks(nNF, TB), TB, 0, stream>>>(bufC, b1, bufA);

  // ---- GATConv 2 ----
  gemm_wmma_kernel<F1><<<NT / 64, 128, 0, stream>>>(bufA, W2, bufB);
  dots_kernel<<<blks(nNH, TB), TB, 0, stream>>>(bufB, as2, ad2, asrc, adst);
  fill_kernel<<<blks(nNH, TB), TB, 0, stream>>>(mbuf, -3.0e38f, (int)nNH);
  fill_kernel<<<blks(nNH, TB), TB, 0, stream>>>(sbuf, 0.f, (int)nNH);
  fill_kernel<<<blks(nNF, TB), TB, 0, stream>>>(bufC, 0.f, (int)nNF);
  edge_max_kernel<<<blks(nEH, TB), TB, 0, stream>>>(ei, asrc, adst, mbuf);
  edge_expsum_kernel<<<blks(nEH, TB), TB, 0, stream>>>(ei, asrc, adst, mbuf, sbuf);
  edge_scatter_kernel<<<blks(nEH, TB), TB, 0, stream>>>(ei, asrc, adst, mbuf, sbuf, bufB, bufC);
  bias_leaky_kernel<<<blks(nNF, TB), TB, 0, stream>>>(bufC, b2, bufA);

  // ---- pool + MLP head ----
  maxpool_kernel<<<blks((long long)NB * NPOOL * F1, TB), TB, 0, stream>>>(bufA, pool);
  lin_kernel<<<blks((long long)NB * LIN_MID, TB), TB, 0, stream>>>(pool, l1W, l1b, l1o,
                                                                   LIN_IN, LIN_MID, 1);
  lin_kernel<<<blks((long long)NB * 64, TB), TB, 0, stream>>>(l1o, l2W, l2b, l2o,
                                                              LIN_MID, 64, 1);
  clf_kernel<<<1, 32, 0, stream>>>(l2o, cW, cb, out);
}
